// SimCLR_42537356099878
// MI455X (gfx1250) — compile-verified, matrix operand order checked by reading, not run on hardware
//
#include <hip/hip_runtime.h>

typedef __attribute__((ext_vector_type(16))) _Float16 v16h;
typedef __attribute__((ext_vector_type(8)))  float    v8f;

union AF { unsigned u[8]; v16h v; };

#define DEV __device__ __forceinline__

DEV unsigned smix(unsigned x){ x+=0x9e3779b9u; x^=x>>16; x*=0x21f0aaadu; x^=x>>15; x*=0x735a2d97u; x^=x>>15; return x; }
DEV float u01(unsigned h){ return (float)(h>>8) * (1.0f/16777216.0f); }
DEV float clampf(float v,float lo,float hi){ return fminf(fmaxf(v,lo),hi); }

// ---------------------------------------------------------------------------
// Weight packing: f32 [N][K] row-major -> f16 "K-chunked N-major" [K/32][N][32]
// ---------------------------------------------------------------------------
__global__ void k_pack_w(const float* __restrict__ Wsrc, _Float16* __restrict__ dst,
                         int N, int K) {
  int idx = blockIdx.x * 256 + threadIdx.x;
  int total = N * K;
  if (idx >= total) return;
  int n = idx / K, k = idx - n * K;
  dst[((size_t)(k >> 5) * N + n) * 32 + (k & 31)] = (_Float16)Wsrc[idx];
}

// W2 f32 OIHW [64][32][3][3] -> chunked f16 with k = (dy*3+dx)*32 + cin
__global__ void k_pack_conv2(const float* __restrict__ W2, _Float16* __restrict__ dst) {
  int idx = blockIdx.x * 256 + threadIdx.x;
  if (idx >= 64 * 288) return;
  int n = idx / 288, k = idx - n * 288;
  int tap = k >> 5, cin = k & 31;
  int dy = tap / 3, dx = tap % 3;
  int src = ((n * 32 + cin) * 3 + dy) * 3 + dx;
  dst[((size_t)(k >> 5) * 64 + n) * 32 + (k & 31)] = (_Float16)W2[src];
}

// ---------------------------------------------------------------------------
// Fused augment -> conv1+relu+pool -> conv2 (implicit-GEMM WMMA) -> relu+pool
// One block = one (view, image). 8 waves, 256 threads.
// LDS: h1 [197][32] f16 (pixel 196 = zero halo pixel), then w2s/c2 overlay.
// ---------------------------------------------------------------------------
__global__ __launch_bounds__(256)
void k_mega(const float* __restrict__ x, const float* __restrict__ W1,
            const float* __restrict__ b1, const _Float16* __restrict__ w2p,
            const float* __restrict__ b2, _Float16* __restrict__ A2)
{
  __shared__ __align__(16) char smem[51776];
  __shared__ float w1s[288];
  __shared__ float b1s[32];
  _Float16* h1  = (_Float16*)smem;               // [197][32] f16  (12608 B)
  _Float16* w2s = (_Float16*)(smem + 12608);     // [9][64][34] f16 (39168 B)
  _Float16* c2  = (_Float16*)(smem + 12608);     // overlay after GEMM: [196][66]
  float* img  = (float*)(smem + 12608);          // phases A/B only (before w2s)
  float* buf1 = (float*)(smem + 12608 + 3136);
  float* aug  = (float*)(smem + 12608 + 6272);

  const int tid = threadIdx.x;
  const int lane = tid & 31, w = tid >> 5;
  const int lane15 = lane & 15;
  const int hi8 = (lane >= 16) ? 8 : 0;
  const int view = blockIdx.x >> 12;             // 4096 images per view
  const int b = blockIdx.x & 4095;
  const float* xi = x + (size_t)b * 784;

  for (int p = tid; p < 784; p += 256) img[p] = xi[p];
  for (int p = tid; p < 288; p += 256) w1s[p] = W1[p];
  if (tid < 32) b1s[tid] = b1[tid];
  if (tid < 16) ((unsigned*)h1)[196 * 16 + tid] = 0u;   // zero halo pixel

  // per-image augment params (deterministic hash; all threads compute same)
  unsigned seed = (unsigned)view * 0xA511E9B3u + (unsigned)b * 0x9E3779B9u + 42u;
  float u[10];
  #pragma unroll
  for (int i = 0; i < 10; ++i) u[i] = u01(smix(seed + (unsigned)i * 0x632BE5ABu));
  float crop = floorf(28.0f * (0.7f + 0.3f * u[0]));
  float top  = floorf((28.0f - crop) * u[1]);
  float left = floorf((28.0f - crop) * u[2]);
  bool flip = u[3] < 0.5f;
  bool rot  = u[4] < 0.8f;
  float rad = (20.0f * u[5] - 10.0f) * 0.017453292519943295f;
  bool cj = u[6] < 0.8f;
  float bb = 0.2f * u[7] - 0.1f;
  float cc = 0.8f + 0.4f * u[8];
  bool na = u[9] < 0.5f;

  __syncthreads();

  // ---- crop-resize (clamp gather) + flip ----
  for (int p = tid; p < 784; p += 256) {
    int j = p / 28, i = p % 28;
    float sy = clampf(((float)j + 0.5f) * crop * (1.0f/28.0f) - 0.5f, 0.0f, crop - 1.0f) + top;
    float sx = clampf(((float)i + 0.5f) * crop * (1.0f/28.0f) - 0.5f, 0.0f, crop - 1.0f) + left;
    float y0 = floorf(sy), x0 = floorf(sx);
    float wy = sy - y0, wx = sx - x0;
    int yi = (int)y0, xi0 = (int)x0;
    auto g = [&](int a, int bq) -> float {
      int ac = a < 0 ? 0 : (a > 27 ? 27 : a);
      int bc = bq < 0 ? 0 : (bq > 27 ? 27 : bq);
      return img[ac * 28 + bc];
    };
    float v = g(yi, xi0) * (1-wy) * (1-wx) + g(yi, xi0+1) * (1-wy) * wx
            + g(yi+1, xi0) * wy * (1-wx) + g(yi+1, xi0+1) * wy * wx;
    buf1[j * 28 + (flip ? 27 - i : i)] = v;
  }
  __syncthreads();

  // ---- rotate (zero-pad) + jitter + noise + normalize ----
  for (int p = tid; p < 784; p += 256) {
    int j = p / 28, i = p % 28;
    float v;
    if (rot) {
      float Yn = (2.0f*j + 1.0f) * (1.0f/28.0f) - 1.0f;
      float Xn = (2.0f*i + 1.0f) * (1.0f/28.0f) - 1.0f;
      float cs = cosf(rad), sn = sinf(rad);
      float xs = cs * Xn - sn * Yn, ys = sn * Xn + cs * Yn;
      float ix = ((xs + 1.0f) * 28.0f - 1.0f) * 0.5f;
      float iy = ((ys + 1.0f) * 28.0f - 1.0f) * 0.5f;
      float y0 = floorf(iy), x0 = floorf(ix);
      float wy = iy - y0, wx = ix - x0;
      int yi = (int)y0, xi0 = (int)x0;
      // load-from-clamped-address first, select on value (no predicated loads)
      auto g = [&](int a, int bq) -> float {
        bool valid = ((unsigned)a < 28u) && ((unsigned)bq < 28u);
        int ac = a < 0 ? 0 : (a > 27 ? 27 : a);
        int bc = bq < 0 ? 0 : (bq > 27 ? 27 : bq);
        float t = buf1[ac * 28 + bc];
        return valid ? t : 0.0f;
      };
      v = g(yi, xi0) * (1-wy) * (1-wx) + g(yi, xi0+1) * (1-wy) * wx
        + g(yi+1, xi0) * wy * (1-wx) + g(yi+1, xi0+1) * wy * wx;
    } else v = buf1[p];
    if (cj) v = clampf((clampf(v + bb, 0.0f, 1.0f) - 0.5f) * cc + 0.5f, 0.0f, 1.0f);
    if (na) {
      unsigned hA = smix(seed ^ (((unsigned)p * 0x85EBCA6Bu) + 0x12D687u));
      unsigned hB = smix((seed + 0x94D049BBu) ^ ((unsigned)p * 0xC2B2AE35u));
      float r = sqrtf(-2.0f * logf(fmaxf(u01(hA), 1e-7f)));
      float nz = r * cosf(6.2831853f * u01(hB));
      v = clampf(v + 0.05f * nz, 0.0f, 1.0f);
    }
    aug[p] = (v - 0.5f) * 2.0f;
  }
  __syncthreads();

  // ---- conv1 (1->32, 3x3 SAME) + relu + 2x2 maxpool -> h1 NHWC f16 ----
  for (int idx = tid; idx < 6272; idx += 256) {
    int c = idx & 31, pp = idx >> 5;
    int py = pp / 14, px = pp % 14;
    float m4 = 0.0f;
    #pragma unroll
    for (int q = 0; q < 4; ++q) {
      int y = 2*py + (q >> 1), xx = 2*px + (q & 1);
      float s = b1s[c];
      #pragma unroll
      for (int dy = 0; dy < 3; ++dy)
        #pragma unroll
        for (int dx = 0; dx < 3; ++dx) {
          int iy = y + dy - 1, ix = xx + dx - 1;
          bool vld = ((unsigned)iy < 28u) && ((unsigned)ix < 28u);
          int iyc = vld ? iy : 0, ixc = vld ? ix : 0;
          float pv = aug[iyc * 28 + ixc];           // unconditional LDS load
          s += w1s[c*9 + dy*3 + dx] * (vld ? pv : 0.0f);
        }
      s = fmaxf(s, 0.0f);
      m4 = fmaxf(m4, s);
    }
    h1[pp * 32 + c] = (_Float16)m4;
  }
  __syncthreads();

  // ---- stage conv2 weights into LDS (n-stride 34 halves, conflict-free) ----
  {
    const unsigned* w2g = (const unsigned*)w2p;  // [9][64][16] u32 contiguous
    for (int idx = tid; idx < 9216; idx += 256) {
      int kc = idx / 1024;
      int r = idx - kc * 1024;
      int n = r >> 4, jj = r & 15;
      *(unsigned*)&w2s[kc * 2176 + n * 34 + 2 * jj] = w2g[idx];
    }
  }
  __syncthreads();

  // ---- conv2 implicit GEMM: M padded to 16 tiles of 16, N=64, K=288 ----
  // Every wave statically owns tiles {w, w+8}; halo taps redirect to the
  // zeroed dummy pixel 196 -> all LDS loads unconditional, EXEC untouched.
  int kb[8];
  #pragma unroll
  for (int j = 0; j < 8; ++j) kb[j] = ((j & 4) ? 16 : 0) + 2 * (j & 3) + hi8;

  v8f acc[2][4];
  #pragma unroll
  for (int a0 = 0; a0 < 2; ++a0)
    #pragma unroll
    for (int a1 = 0; a1 < 4; ++a1)
      #pragma unroll
      for (int e = 0; e < 8; ++e) acc[a0][a1][e] = 0.0f;

  int py_[2], px_[2];
  #pragma unroll
  for (int tt = 0; tt < 2; ++tt) {
    int t = w + tt * 8;
    int p = t * 16 + lane15;
    if (p > 195) p = 195;
    py_[tt] = p / 14; px_[tt] = p % 14;
  }
  const unsigned* h1u = (const unsigned*)h1;

  #pragma unroll
  for (int kc = 0; kc < 9; ++kc) {
    int dy = kc / 3 - 1, dx = kc % 3 - 1;
    AF af[2];
    #pragma unroll
    for (int tt = 0; tt < 2; ++tt) {
      int iy = py_[tt] + dy, ix = px_[tt] + dx;
      bool inb = ((unsigned)iy < 14u) && ((unsigned)ix < 14u);
      int pidx = inb ? (iy * 14 + ix) : 196;      // 196 = zero pixel
      int base = pidx * 16;
      #pragma unroll
      for (int j = 0; j < 8; ++j) af[tt].u[j] = h1u[base + (kb[j] >> 1)];
    }
    #pragma unroll
    for (int ns = 0; ns < 4; ++ns) {
      AF bf;
      int nb = kc * 2176 + (ns * 16 + lane15) * 34;
      #pragma unroll
      for (int j = 0; j < 8; ++j) bf.u[j] = *(const unsigned*)&w2s[nb + kb[j]];
      #pragma unroll
      for (int tt = 0; tt < 2; ++tt)
        acc[tt][ns] = __builtin_amdgcn_wmma_f32_16x16x32_f16(
            false, af[tt].v, false, bf.v, (short)0, acc[tt][ns], false, false);
    }
  }
  __syncthreads();   // all WMMA reads of w2s done; overlay c2 on that region

  #pragma unroll
  for (int tt = 0; tt < 2; ++tt) {
    int t = w + tt * 8;
    #pragma unroll
    for (int ns = 0; ns < 4; ++ns) {
      int n = ns * 16 + lane15;
      float bias = b2[n];
      #pragma unroll
      for (int r = 0; r < 8; ++r) {
        int m = t * 16 + r + hi8;
        if (m < 196)
          c2[m * 66 + n] = (_Float16)fmaxf(acc[tt][ns][r] + bias, 0.0f);
      }
    }
  }
  __syncthreads();

  // ---- 2x2 maxpool -> flattened row, NCHW flatten order c*49 + y*7 + x ----
  _Float16* outrow = A2 + (size_t)(view * 4096 + b) * 3136;
  for (int idx = tid; idx < 3136; idx += 256) {
    int c = idx / 49, rem = idx % 49;
    int py = rem / 7, px = rem % 7;
    int m0 = (2 * py) * 14 + 2 * px;
    float v0 = (float)c2[m0 * 66 + c];
    float v1 = (float)c2[(m0 + 1) * 66 + c];
    float v2 = (float)c2[(m0 + 14) * 66 + c];
    float v3 = (float)c2[(m0 + 15) * 66 + c];
    outrow[idx] = (_Float16)fmaxf(fmaxf(v0, v1), fmaxf(v2, v3));
  }
}

// ---------------------------------------------------------------------------
// Generic WMMA GEMM: C[M,N] = act(A[M,K] * Bw + bias). Block tile 128x64,
// 8 waves each 16x64. A f16 row-major, Bw f16 chunked [K/32][N][32].
// ---------------------------------------------------------------------------
template<bool RELU, bool OUTF16>
__global__ __launch_bounds__(256)
void k_gemm(const _Float16* __restrict__ A, const _Float16* __restrict__ Bw,
            const float* __restrict__ bias, void* __restrict__ out,
            int M, int N, int K)
{
  __shared__ __align__(16) _Float16 As[128 * 36];
  __shared__ __align__(16) _Float16 Bs[64 * 34];
  const int tid = threadIdx.x;
  const int lane = tid & 31, w = tid >> 5;
  const int lane15 = lane & 15;
  const int hi8 = (lane >= 16) ? 8 : 0;
  const int m0 = blockIdx.y * 128;
  const int n0 = blockIdx.x * 64;

  int kb[8];
  #pragma unroll
  for (int j = 0; j < 8; ++j) kb[j] = ((j & 4) ? 16 : 0) + 2 * (j & 3) + hi8;

  v8f acc[4];
  #pragma unroll
  for (int ns = 0; ns < 4; ++ns)
    #pragma unroll
    for (int e = 0; e < 8; ++e) acc[ns][e] = 0.0f;

  const int nk = K >> 5;
  const int rA = tid >> 1;
  const int oA = (tid & 1) * 16;
  const int nB = tid >> 2;
  const int hB = (tid & 3) * 8;

  for (int kc = 0; kc < nk; ++kc) {
    // stage 128x32 A tile (row stride 36 halves -> conflict-free frag reads)
    const _Float16* ag = A + (size_t)(m0 + rA) * K + kc * 32 + oA;
    #pragma unroll
    for (int j = 0; j < 8; ++j)
      *(unsigned*)&As[rA * 36 + oA + 2 * j] = *(const unsigned*)(ag + 2 * j);
    // stage 64-col B chunk (n stride 34 halves)
    const _Float16* bg = Bw + ((size_t)kc * N + n0 + nB) * 32 + hB;
    #pragma unroll
    for (int j = 0; j < 4; ++j)
      *(unsigned*)&Bs[nB * 34 + hB + 2 * j] = *(const unsigned*)(bg + 2 * j);
    if (kc + 1 < nk)
      __builtin_prefetch(A + (size_t)(m0 + rA) * K + (kc + 1) * 32 + oA, 0, 0);
    __syncthreads();

    AF af;
    #pragma unroll
    for (int j = 0; j < 8; ++j)
      af.u[j] = *(const unsigned*)&As[(w * 16 + lane15) * 36 + kb[j]];
    #pragma unroll
    for (int ns = 0; ns < 4; ++ns) {
      AF bf;
      #pragma unroll
      for (int j = 0; j < 8; ++j)
        bf.u[j] = *(const unsigned*)&Bs[(ns * 16 + lane15) * 34 + kb[j]];
      acc[ns] = __builtin_amdgcn_wmma_f32_16x16x32_f16(
          false, af.v, false, bf.v, (short)0, acc[ns], false, false);
    }
    __syncthreads();
  }

  #pragma unroll
  for (int ns = 0; ns < 4; ++ns) {
    int n = n0 + ns * 16 + lane15;
    float bv = bias[n];
    #pragma unroll
    for (int r = 0; r < 8; ++r) {
      int m = m0 + w * 16 + r + hi8;
      float v = acc[ns][r] + bv;
      if (RELU) v = fmaxf(v, 0.0f);
      if (OUTF16) ((_Float16*)out)[(size_t)m * N + n] = (_Float16)v;
      else        ((float*)out)[(size_t)m * N + n] = v;
    }
  }
}

// ---------------------------------------------------------------------------
extern "C" void kernel_launch(void* const* d_in, const int* in_sizes, int n_in,
                              void* d_out, int out_size, void* d_ws, size_t ws_size,
                              hipStream_t stream)
{
  const float* x   = (const float*)d_in[0];
  const float* W1  = (const float*)d_in[1];
  const float* b1  = (const float*)d_in[2];
  const float* W2  = (const float*)d_in[3];
  const float* b2  = (const float*)d_in[4];
  const float* Wf  = (const float*)d_in[5];
  const float* bf  = (const float*)d_in[6];
  const float* Wp1 = (const float*)d_in[7];
  const float* bp1 = (const float*)d_in[8];
  const float* Wp2 = (const float*)d_in[9];
  const float* bp2 = (const float*)d_in[10];
  (void)in_sizes; (void)n_in; (void)out_size; (void)ws_size;

  char* ws = (char*)d_ws;
  size_t off = 0;
  auto take = [&](size_t bytes) -> char* {
    char* p = ws + off; off = (off + bytes + 255) & ~(size_t)255; return p;
  };
  _Float16* wc2p = (_Float16*)take((size_t)288 * 64 * 2);
  _Float16* wfp  = (_Float16*)take((size_t)3136 * 512 * 2);
  _Float16* wp1p = (_Float16*)take((size_t)512 * 512 * 2);
  _Float16* wp2p = (_Float16*)take((size_t)512 * 128 * 2);
  _Float16* A2   = (_Float16*)take((size_t)8192 * 3136 * 2);  // both views flat
  _Float16* hv   = (_Float16*)take((size_t)8192 * 512 * 2);
  _Float16* p1v  = (_Float16*)take((size_t)8192 * 512 * 2);

  k_pack_conv2<<<(64 * 288 + 255) / 256, 256, 0, stream>>>(W2, wc2p);
  k_pack_w<<<(512 * 3136 + 255) / 256, 256, 0, stream>>>(Wf, wfp, 512, 3136);
  k_pack_w<<<(512 * 512 + 255) / 256, 256, 0, stream>>>(Wp1, wp1p, 512, 512);
  k_pack_w<<<(128 * 512 + 255) / 256, 256, 0, stream>>>(Wp2, wp2p, 128, 512);

  k_mega<<<8192, 256, 0, stream>>>(x, W1, b1, wc2p, b2, A2);

  // encoder FC (no activation), projector hidden (relu), projector out (f32)
  k_gemm<false, true ><<<dim3(8, 64), 256, 0, stream>>>(A2, wfp, bf, hv, 8192, 512, 3136);
  k_gemm<true,  true ><<<dim3(8, 64), 256, 0, stream>>>(hv, wp1p, bp1, p1v, 8192, 512, 512);
  k_gemm<false, false><<<dim3(2, 64), 256, 0, stream>>>(p1v, wp2p, bp2, d_out, 8192, 128, 512);
}
